// ReformerDecoder_51479478010645
// MI455X (gfx1250) — compile-verified, matrix operand order checked by reading, not run on hardware
//
#include <hip/hip_runtime.h>
#include <hip/hip_bf16.h>
#include <math.h>

// ---------------- model constants ----------------
constexpr int B = 2, L = 2048, D = 512, H = 8, NL = 6, FF = 2048, V = 32000;
constexpr int DH = 64, CHUNK = 64, NC = L / CHUNK, NHASH = 2, NB2 = 16;
constexpr float NEG = -1e9f, SELF_PEN = -1e5f;

#define USE_ASYNC_COPY 1

// ---------------- WMMA types/helpers ----------------
typedef __attribute__((ext_vector_type(16))) __bf16 v16bf;
typedef __attribute__((ext_vector_type(8)))  float  v8f;

__device__ __forceinline__ unsigned short f2bf_bits(float f) {
    union { float f; unsigned u; } c; c.f = f;
    unsigned u = c.u;
    u += 0x7fffu + ((u >> 16) & 1u);          // round-to-nearest-even
    return (unsigned short)(u >> 16);
}
__device__ __forceinline__ __bf16 bits2bf(unsigned short s) {
    union { unsigned short s; __bf16 b; } c; c.s = s; return c.b;
}
// per-lane K index pattern for 16-bit A/B fragments (16x16x32):
// lanes 0-15: K = {0..7, 16..23}; lanes 16-31: K = {8..15, 24..31}, pair-packed.
__device__ __forceinline__ int frag_k(int tp, bool hi) {
    return 2 * tp + (tp < 4 ? (hi ? 8 : 0) : (hi ? 16 : 8));
}
// fragment from [row][k] layout (adjacent k pairs contiguous)
__device__ __forceinline__ v16bf load_frag_rowk(const unsigned short* row, bool hi) {
    v16bf f;
#pragma unroll
    for (int tp = 0; tp < 8; ++tp) {
        int k = frag_k(tp, hi);
        f[2 * tp]     = bits2bf(row[k]);
        f[2 * tp + 1] = bits2bf(row[k + 1]);
    }
    return f;
}
// B fragment from natural [k][n] layout (stride = tile width in n)
__device__ __forceinline__ v16bf load_frag_kn(const unsigned short* base, int n, int ldn, bool hi) {
    v16bf f;
#pragma unroll
    for (int tp = 0; tp < 8; ++tp) {
        int k = frag_k(tp, hi);
        f[2 * tp]     = bits2bf(base[k * ldn + n]);
        f[2 * tp + 1] = bits2bf(base[(k + 1) * ldn + n]);
    }
    return f;
}

// ---------------- fp32 -> bf16 bulk convert ----------------
__global__ __launch_bounds__(256) void convert_bf16_kernel(const float* __restrict__ in,
                                                           unsigned short* __restrict__ out,
                                                           int n) {
    int i = (blockIdx.x * 256 + threadIdx.x) * 4;
    if (i + 3 < n) {
        float4 v = *(const float4*)(in + i);
        out[i]     = f2bf_bits(v.x);
        out[i + 1] = f2bf_bits(v.y);
        out[i + 2] = f2bf_bits(v.z);
        out[i + 3] = f2bf_bits(v.w);
    }
}

// ---------------- embedding (shift-right gather + pos) ----------------
__global__ __launch_bounds__(256) void embed_kernel(const int* __restrict__ ids,
                                                    const float* __restrict__ emb,
                                                    const float* __restrict__ pos,
                                                    float* __restrict__ x) {
    int bl = blockIdx.x, b = bl / L, l = bl % L;
    int id = (l == 0) ? 0 : ids[b * L + l - 1];
    int d = threadIdx.x;
    const float* e = emb + (size_t)id * D;
    const float* p = pos + (size_t)l * D;
    float* o = x + (size_t)bl * D;
    o[d]       = e[d]       + p[d];
    o[d + 256] = e[d + 256] + p[d + 256];
}

// ---------------- layernorm: one block per row, bf16 output ----------------
__global__ __launch_bounds__(256) void layernorm_kernel(const float* __restrict__ in,
                                                        const float* __restrict__ g,
                                                        const float* __restrict__ bta,
                                                        unsigned short* __restrict__ out) {
    __shared__ float s1[256], s2[256];
    int row = blockIdx.x, t = threadIdx.x;
    const float* r = in + (size_t)row * D;
    float a = r[t], c = r[t + 256];
    s1[t] = a + c; s2[t] = a * a + c * c;
    __syncthreads();
    for (int o = 128; o > 0; o >>= 1) {
        if (t < o) { s1[t] += s1[t + o]; s2[t] += s2[t + o]; }
        __syncthreads();
    }
    float mean = s1[0] * (1.f / D);
    float var  = s2[0] * (1.f / D) - mean * mean;
    float inv  = rsqrtf(var + 1e-6f);
    unsigned short* w = out + (size_t)row * D;
    w[t]       = f2bf_bits((a - mean) * inv * g[t]       + bta[t]);
    w[t + 256] = f2bf_bits((c - mean) * inv * g[t + 256] + bta[t + 256]);
}

// ---------------- async tile staging (one b128 per lane per matrix) ----------------
__device__ __forceinline__ void issue_tile_async(const unsigned short* __restrict__ A,
                                                 const unsigned short* __restrict__ Bw,
                                                 unsigned short* Asb, unsigned short* Bsb,
                                                 int tid, int m0, int n0, int k0,
                                                 int Kdim, int Ndim) {
    // A tile: 64 rows x 32 bf16 = 4KB = 256 lanes x 16B
    int am = tid >> 2, aseg = tid & 3;
    const unsigned short* ga = A + (size_t)(m0 + am) * Kdim + k0 + aseg * 8;
    unsigned al = (unsigned)(unsigned long long)(Asb + am * 32 + aseg * 8);
    asm volatile("global_load_async_to_lds_b128 %0, %1, off" :: "v"(al), "v"(ga) : "memory");
    // B tile: 32 rows x 64 bf16 = 4KB
    int bk = tid >> 3, bseg = tid & 7;
    const unsigned short* gb = Bw + (size_t)(k0 + bk) * Ndim + n0 + bseg * 8;
    unsigned bl = (unsigned)(unsigned long long)(Bsb + bk * 64 + bseg * 8);
    asm volatile("global_load_async_to_lds_b128 %0, %1, off" :: "v"(bl), "v"(gb) : "memory");
}

// ---------------- bf16-in / bf16-WMMA GEMM, fp32 accumulate ----------------
// Double-buffered LDS; async copies pipelined against WMMA via partial ASYNCcnt waits.
// Cf/Cb[M,N] (+= Cf if RESID) relu( A[M,K] @ Bw[K,N] + bias )
template <int BIAS, int RELU, int RESID, int OUTBF>
__global__ __launch_bounds__(256) void gemm_bf16_wmma(const unsigned short* __restrict__ A,
                                                      const unsigned short* __restrict__ Bw,
                                                      const float* __restrict__ bias,
                                                      float* __restrict__ Cf,
                                                      unsigned short* __restrict__ Cb,
                                                      int Mdim, int Ndim, int Kdim) {
    __shared__ unsigned short As[2][64 * 32];   // [m][k]
    __shared__ unsigned short Bs[2][32 * 64];   // [k][n] natural layout (async-copy friendly)
    const int tid = threadIdx.x, lane = tid & 31, wave = tid >> 5;
    const bool hi = lane >= 16; const int l16 = lane & 15;
    const int m0 = blockIdx.y * 64, n0 = blockIdx.x * 64;
    const int wr = wave & 3, wc = wave >> 2;          // 4x2 waves -> 64x64 tile
    v8f acc0 = {}; v8f acc1 = {};
    const int nk = Kdim >> 5;
#if USE_ASYNC_COPY
    issue_tile_async(A, Bw, As[0], Bs[0], tid, m0, n0, 0, Kdim, Ndim);
#endif
    for (int ki = 0; ki < nk; ++ki) {
        const int cur = ki & 1;
#if USE_ASYNC_COPY
        if (ki + 1 < nk) {
            // stream next tile into the other buffer while we compute on this one
            issue_tile_async(A, Bw, As[cur ^ 1], Bs[cur ^ 1], tid, m0, n0, (ki + 1) * 32, Kdim, Ndim);
            if (ki + 2 < nk) {   // warm caches for the tile after next
                __builtin_prefetch(A + (size_t)(m0 + (tid >> 2)) * Kdim + (ki + 2) * 32, 0, 1);
                __builtin_prefetch(Bw + (size_t)((ki + 2) * 32 + (tid >> 3)) * Ndim + n0, 0, 1);
            }
            asm volatile("s_wait_asynccnt 2" ::: "memory");   // oldest tile (2 loads) landed
        } else {
            asm volatile("s_wait_asynccnt 0" ::: "memory");
        }
#else
        {
            unsigned* Asu = (unsigned*)As[cur];
            const unsigned* ga = (const unsigned*)A;
            for (int i = tid; i < 64 * 16; i += 256) {
                int m = i >> 4, ku = i & 15;
                Asu[m * 16 + ku] = ga[((size_t)(m0 + m) * Kdim + ki * 32) / 2 + ku];
            }
            unsigned* Bsu = (unsigned*)Bs[cur];
            const unsigned* gb = (const unsigned*)Bw;
            for (int i = tid; i < 32 * 32; i += 256) {
                int k = i >> 5, nu = i & 31;
                Bsu[k * 32 + nu] = gb[((size_t)(ki * 32 + k) * Ndim + n0) / 2 + nu];
            }
        }
#endif
        __syncthreads();
        v16bf af  = load_frag_rowk(&As[cur][(wr * 16 + l16) * 32], hi);
        v16bf bf0 = load_frag_kn(Bs[cur], wc * 32 + l16, 64, hi);
        v16bf bf1 = load_frag_kn(Bs[cur], wc * 32 + 16 + l16, 64, hi);
        acc0 = __builtin_amdgcn_wmma_f32_16x16x32_bf16(false, af, false, bf0, (short)0, acc0, false, false);
        acc1 = __builtin_amdgcn_wmma_f32_16x16x32_bf16(false, af, false, bf1, (short)0, acc1, false, false);
        __syncthreads();    // done reading buf 'cur'; it may be overwritten next+1 iteration
    }
#pragma unroll
    for (int s = 0; s < 2; ++s) {
        v8f acc = s ? acc1 : acc0;
        int ncol = n0 + wc * 32 + s * 16 + l16;
#pragma unroll
        for (int r = 0; r < 8; ++r) {
            int row = m0 + wr * 16 + r + (hi ? 8 : 0);
            float val = acc[r];
            if (BIAS) val += bias[ncol];
            if (RELU) val = fmaxf(val, 0.f);
            size_t idx = (size_t)row * Ndim + ncol;
            if (RESID) val += Cf[idx];
            if (OUTBF) Cb[idx] = f2bf_bits(val);
            else       Cf[idx] = val;
        }
    }
}

// ---------------- LSH bucketing + stable sort (bitonic on bucket*L+pos) ----------------
__global__ __launch_bounds__(1024) void lsh_sort_kernel(const float* __restrict__ qk,
                                                        const float* __restrict__ rotL,
                                                        int* __restrict__ perm) {
    __shared__ float rotS[DH * NB2];
    __shared__ unsigned int keys[L];
    int bid = blockIdx.x;                          // ((b*H+h)*NHASH+hs)
    int b = bid / (H * NHASH);
    int h = (bid / NHASH) % H;
    int hs = bid % NHASH;
    int t = threadIdx.x;
    rotS[t] = rotL[((size_t)(h * NHASH + hs)) * DH * NB2 + t];   // DH*NB2 == 1024
    __syncthreads();
    for (int l = t; l < L; l += 1024) {
        const float* q = qk + (((size_t)b * L + l) * H + h) * DH;
        const float4* q4 = (const float4*)q;
        float qv[DH];
#pragma unroll
        for (int i = 0; i < 16; ++i) {
            float4 v4 = q4[i];
            qv[4 * i] = v4.x; qv[4 * i + 1] = v4.y; qv[4 * i + 2] = v4.z; qv[4 * i + 3] = v4.w;
        }
        float rv[NB2];
#pragma unroll
        for (int nb = 0; nb < NB2; ++nb) rv[nb] = 0.f;
        for (int d = 0; d < DH; ++d) {
            float qd = qv[d];
            const float* rr = &rotS[d * NB2];
#pragma unroll
            for (int nb = 0; nb < NB2; ++nb) rv[nb] += qd * rr[nb];
        }
        float best = -1e30f; int bb = 0;
#pragma unroll
        for (int j = 0; j < 2 * NB2; ++j) {       // argmax over [rot, -rot], first max wins
            float vv = (j < NB2) ? rv[j] : -rv[j - NB2];
            if (vv > best) { best = vv; bb = j; }
        }
        keys[l] = ((unsigned)bb << 11) | (unsigned)l;   // bucket*2048 + pos (stable)
    }
    __syncthreads();
    for (unsigned k = 2; k <= (unsigned)L; k <<= 1) {
        for (unsigned j = k >> 1; j > 0; j >>= 1) {
            for (unsigned i = t; i < (unsigned)L; i += 1024) {
                unsigned ixj = i ^ j;
                if (ixj > i) {
                    unsigned a = keys[i], c = keys[ixj];
                    bool asc = ((i & k) == 0);
                    if ((asc && a > c) || (!asc && a < c)) { keys[i] = c; keys[ixj] = a; }
                }
            }
            __syncthreads();
        }
    }
    int* po = perm + (size_t)bid * L;
    for (int i = t; i < L; i += 1024) po[i] = (int)(keys[i] & (L - 1));
}

// ---------------- chunked LSH attention (one block per (b,h,hash,chunk)) ----------------
__global__ __launch_bounds__(256) void lsh_attn_kernel(const float* __restrict__ qk,
                                                       const float* __restrict__ vbuf,
                                                       const int* __restrict__ perm,
                                                       float* __restrict__ obuf,
                                                       float* __restrict__ lsebuf) {
    // Sbuf (32KB f32) is aliased: first 8KB = Qs (bf16 64x64), next 16KB = Ks (bf16 128x64)
    // during phases 1-2; becomes scores S[64][128] after a barrier.
    __shared__ float Sbuf[64 * 128];
    __shared__ unsigned short Vsm[128 * 64];      // V rows (bf16, [key][d])
    __shared__ float lseS[64];
    __shared__ int ctS[64];
    __shared__ int btS[128];
    unsigned short* Qs  = (unsigned short*)Sbuf;             // [64][64]
    unsigned short* Ksm = ((unsigned short*)Sbuf) + 64 * 64; // [128][64]

    int bid = blockIdx.x;
    int c  = bid % NC;
    int bhh = bid / NC;                     // ((b*H+h)*NHASH+hs)
    int b  = bhh / (H * NHASH);
    int h  = (bhh / NHASH) % H;
    int t = threadIdx.x, lane = t & 31, wave = t >> 5;
    bool hi = lane >= 16; int l16 = lane & 15;

    // ---- phase 1: gather sorted Q/K/V rows, length-normalize keys ----
    if (t < 128) {
        int j = t;
        int chunkj = (j < 64) ? c : (c + NC - 1) % NC;      // own chunk then look-back chunk
        int sp = perm[(size_t)bhh * L + chunkj * 64 + (j & 63)];
        btS[j] = sp;
        const float* qrow = qk   + (((size_t)b * L + sp) * H + h) * DH;
        const float* vrow = vbuf + (((size_t)b * L + sp) * H + h) * DH;
        const float4* q4 = (const float4*)qrow;
        float qv[DH]; float ss = 0.f;
#pragma unroll
        for (int i = 0; i < 16; ++i) {
            float4 v4 = q4[i];
            qv[4 * i] = v4.x; qv[4 * i + 1] = v4.y; qv[4 * i + 2] = v4.z; qv[4 * i + 3] = v4.w;
            ss += v4.x * v4.x + v4.y * v4.y + v4.z * v4.z + v4.w * v4.w;
        }
        float inv = 1.f / (sqrtf(ss) + 1e-6f);
#pragma unroll
        for (int d = 0; d < DH; ++d) Ksm[j * 64 + d] = f2bf_bits(qv[d] * inv);
        if (j < 64) {
            ctS[j] = sp;
#pragma unroll
            for (int d = 0; d < DH; ++d) Qs[j * 64 + d] = f2bf_bits(qv[d]);
        }
        const float4* v4p = (const float4*)vrow;
#pragma unroll
        for (int i = 0; i < 16; ++i) {
            float4 w4 = v4p[i];
            Vsm[j * 64 + 4 * i]     = f2bf_bits(w4.x);
            Vsm[j * 64 + 4 * i + 1] = f2bf_bits(w4.y);
            Vsm[j * 64 + 4 * i + 2] = f2bf_bits(w4.z);
            Vsm[j * 64 + 4 * i + 3] = f2bf_bits(w4.w);
        }
    }
    __syncthreads();

    // ---- phase 2: S = Q @ K^T (WMMA), scale + causal/self mask in registers ----
    int tr = wave >> 1;                 // 4 row tiles
    int tc0 = (wave & 1) * 4;           // 8 col tiles, 4 per wave
    v8f zero = {};
    v8f accS[4]; accS[0] = zero; accS[1] = zero; accS[2] = zero; accS[3] = zero;
#pragma unroll
    for (int k0 = 0; k0 < DH; k0 += 32) {
        v16bf af = load_frag_rowk(&Qs[(tr * 16 + l16) * 64 + k0], hi);
#pragma unroll
        for (int j = 0; j < 4; ++j) {
            v16bf bfr = load_frag_rowk(&Ksm[((tc0 + j) * 16 + l16) * 64 + k0], hi);
            accS[j] = __builtin_amdgcn_wmma_f32_16x16x32_bf16(false, af, false, bfr, (short)0, accS[j], false, false);
        }
    }
#pragma unroll
    for (int j = 0; j < 4; ++j)
#pragma unroll
        for (int r = 0; r < 8; ++r) {
            int row = tr * 16 + r + (hi ? 8 : 0);
            int col = (tc0 + j) * 16 + l16;
            int ct = ctS[row], bt = btS[col];
            float sv = accS[j][r] * 0.125f;                    // 1/sqrt(64)
            sv = (ct == bt) ? SELF_PEN : ((ct >= bt) ? sv : NEG);
            accS[j][r] = sv;
        }
    __syncthreads();                 // all Qs/Ks reads done; Sbuf region now reusable
#pragma unroll
    for (int j = 0; j < 4; ++j)
#pragma unroll
        for (int r = 0; r < 8; ++r) {
            int row = tr * 16 + r + (hi ? 8 : 0);
            int col = (tc0 + j) * 16 + l16;
            Sbuf[row * 128 + col] = accS[j][r];
        }
    __syncthreads();

    // ---- phase 3: row LSE (wave shuffles) then P = exp(S - lse) in place ----
#pragma unroll
    for (int rr = 0; rr < 8; ++rr) {
        int row = wave * 8 + rr;
        float x0 = Sbuf[row * 128 + lane];
        float x1 = Sbuf[row * 128 + lane + 32];
        float x2 = Sbuf[row * 128 + lane + 64];
        float x3 = Sbuf[row * 128 + lane + 96];
        float mx = fmaxf(fmaxf(x0, x1), fmaxf(x2, x3));
#pragma unroll
        for (int off = 16; off > 0; off >>= 1) mx = fmaxf(mx, __shfl_xor(mx, off, 32));
        float sm = expf(x0 - mx) + expf(x1 - mx) + expf(x2 - mx) + expf(x3 - mx);
#pragma unroll
        for (int off = 16; off > 0; off >>= 1) sm += __shfl_xor(sm, off, 32);
        float lse = mx + logf(sm);
        if (lane == 0) {
            lseS[row] = lse;
            lsebuf[(size_t)bhh * L + ctS[row]] = lse;          // scatter to original pos
        }
    }
    __syncthreads();
    for (int i = t; i < 64 * 128; i += 256) {
        int row = i >> 7;
        Sbuf[i] = expf(Sbuf[i] - lseS[row]);
    }
    __syncthreads();

    // ---- phase 4: O = P @ V (WMMA), scatter to original positions ----
    int otr = wave >> 1;                 // 4 row tiles
    int otc0 = (wave & 1) * 2;           // 4 col tiles, 2 per wave
    v8f accO[2]; accO[0] = zero; accO[1] = zero;
#pragma unroll
    for (int k0 = 0; k0 < 128; k0 += 32) {
        v16bf af;
        {
            const float* prow = &Sbuf[(otr * 16 + l16) * 128 + k0];
#pragma unroll
            for (int tp = 0; tp < 8; ++tp) {
                int k = frag_k(tp, hi);
                af[2 * tp]     = bits2bf(f2bf_bits(prow[k]));
                af[2 * tp + 1] = bits2bf(f2bf_bits(prow[k + 1]));
            }
        }
#pragma unroll
        for (int j = 0; j < 2; ++j) {
            int n = (otc0 + j) * 16 + l16;
            v16bf bfr = load_frag_kn(Vsm + k0 * 64, n, 64, hi);
            accO[j] = __builtin_amdgcn_wmma_f32_16x16x32_bf16(false, af, false, bfr, (short)0, accO[j], false, false);
        }
    }
    size_t obase = (size_t)bhh * L * DH;
#pragma unroll
    for (int j = 0; j < 2; ++j)
#pragma unroll
        for (int r = 0; r < 8; ++r) {
            int row = otr * 16 + r + (hi ? 8 : 0);
            int col = (otc0 + j) * 16 + l16;
            obuf[obase + (size_t)ctS[row] * DH + col] = accO[j][r];
        }
}

// ---------------- combine the two hash rounds by softmax(lse), bf16 out ----------------
__global__ __launch_bounds__(512) void combine_kernel(const float* __restrict__ obuf,
                                                      const float* __restrict__ lsebuf,
                                                      unsigned short* __restrict__ attn) {
    int bl = blockIdx.x, b = bl / L, l = bl % L;
    int t = threadIdx.x, h = t >> 6, d = t & 63;
    size_t base = ((size_t)(b * H + h)) * NHASH;
    float l0 = lsebuf[(base + 0) * L + l];
    float l1 = lsebuf[(base + 1) * L + l];
    float mx = fmaxf(l0, l1);
    float e0 = expf(l0 - mx), e1 = expf(l1 - mx);
    float w0 = e0 / (e0 + e1);
    float o0 = obuf[((base + 0) * L + l) * DH + d];
    float o1 = obuf[((base + 1) * L + l) * DH + d];
    attn[(size_t)bl * D + h * DH + d] = f2bf_bits(w0 * o0 + (1.f - w0) * o1);
}

// ---------------- host orchestration ----------------
extern "C" void kernel_launch(void* const* d_in, const int* in_sizes, int n_in,
                              void* d_out, int out_size, void* d_ws, size_t ws_size,
                              hipStream_t stream) {
    (void)in_sizes; (void)n_in; (void)out_size; (void)ws_size;
    const int*   inputs = (const int*)d_in[0];
    const float* embed  = (const float*)d_in[1];
    const float* pose   = (const float*)d_in[2];
    const float* ln1_s  = (const float*)d_in[3];
    const float* ln1_b  = (const float*)d_in[4];
    const float* wqk    = (const float*)d_in[5];
    const float* wv     = (const float*)d_in[6];
    const float* wo     = (const float*)d_in[7];
    const float* ln2_s  = (const float*)d_in[8];
    const float* ln2_b  = (const float*)d_in[9];
    const float* w1     = (const float*)d_in[10];
    const float* b1     = (const float*)d_in[11];
    const float* w2     = (const float*)d_in[12];
    const float* b2     = (const float*)d_in[13];
    const float* lnf_s  = (const float*)d_in[14];
    const float* lnf_b  = (const float*)d_in[15];
    const float* wout   = (const float*)d_in[16];
    const float* bout   = (const float*)d_in[17];
    const float* rot    = (const float*)d_in[18];
    float* out = (float*)d_out;

    const size_t nBLD = (size_t)B * L * D;              // 2,097,152
    float* ws    = (float*)d_ws;
    float* x     = ws;                                  // [B*L, D] f32
    float* qkb   = ws + 1 * nBLD;
    float* vb    = ws + 2 * nBLD;
    float* obuf  = ws + 3 * nBLD;                       // 2*nBLD
    float* lseb  = ws + 5 * nBLD;                       // B*H*NHASH*L = 65536
    int*   permb = (int*)(lseb + (size_t)B * H * NHASH * L);
    unsigned short* us0   = (unsigned short*)(ws + 5 * nBLD + 2 * (size_t)B * H * NHASH * L);
    unsigned short* hb    = us0;                        // LN output, bf16 [B*L, D]
    unsigned short* attnb = us0 + 1 * nBLD;             // combine out, bf16
    unsigned short* ff1b  = us0 + 2 * nBLD;             // FFN hidden, bf16 (4*nBLD)
    unsigned short* wbqk  = us0 + 6 * nBLD;             // per-layer weight staging (bf16)
    unsigned short* wbv   = wbqk + (size_t)D * D;
    unsigned short* wbo   = wbv  + (size_t)D * D;
    unsigned short* wb1   = wbo  + (size_t)D * D;
    unsigned short* wb2   = wb1  + (size_t)D * FF;
    unsigned short* wbout = wb2  + (size_t)FF * D;      // [D, V] bf16 (16.4M)

    const int ROWS = B * L;                              // 4096
    convert_bf16_kernel<<<(D * V) / 1024, 256, 0, stream>>>(wout, wbout, D * V);
    embed_kernel<<<ROWS, 256, 0, stream>>>(inputs, embed, pose, x);

    for (int i = 0; i < NL; ++i) {
        convert_bf16_kernel<<<(D * D) / 1024, 256, 0, stream>>>(wqk + (size_t)i * D * D, wbqk, D * D);
        convert_bf16_kernel<<<(D * D) / 1024, 256, 0, stream>>>(wv + (size_t)i * D * D, wbv, D * D);
        convert_bf16_kernel<<<(D * D) / 1024, 256, 0, stream>>>(wo + (size_t)i * D * D, wbo, D * D);
        convert_bf16_kernel<<<(D * FF) / 1024, 256, 0, stream>>>(w1 + (size_t)i * D * FF, wb1, D * FF);
        convert_bf16_kernel<<<(FF * D) / 1024, 256, 0, stream>>>(w2 + (size_t)i * FF * D, wb2, FF * D);

        layernorm_kernel<<<ROWS, 256, 0, stream>>>(x, ln1_s + (size_t)i * D, ln1_b + (size_t)i * D, hb);
        gemm_bf16_wmma<0, 0, 0, 0><<<dim3(D / 64, ROWS / 64), 256, 0, stream>>>(
            hb, wbqk, nullptr, qkb, nullptr, ROWS, D, D);
        gemm_bf16_wmma<0, 0, 0, 0><<<dim3(D / 64, ROWS / 64), 256, 0, stream>>>(
            hb, wbv, nullptr, vb, nullptr, ROWS, D, D);
        lsh_sort_kernel<<<B * H * NHASH, 1024, 0, stream>>>(
            qkb, rot + (size_t)i * H * NHASH * DH * NB2, permb);
        lsh_attn_kernel<<<B * H * NHASH * NC, 256, 0, stream>>>(qkb, vb, permb, obuf, lseb);
        combine_kernel<<<ROWS, 512, 0, stream>>>(obuf, lseb, attnb);
        gemm_bf16_wmma<0, 0, 1, 0><<<dim3(D / 64, ROWS / 64), 256, 0, stream>>>(
            attnb, wbo, nullptr, x, nullptr, ROWS, D, D);
        layernorm_kernel<<<ROWS, 256, 0, stream>>>(x, ln2_s + (size_t)i * D, ln2_b + (size_t)i * D, hb);
        gemm_bf16_wmma<1, 1, 0, 1><<<dim3(FF / 64, ROWS / 64), 256, 0, stream>>>(
            hb, wb1, b1 + (size_t)i * FF, nullptr, ff1b, ROWS, FF, D);
        gemm_bf16_wmma<1, 0, 1, 0><<<dim3(D / 64, ROWS / 64), 256, 0, stream>>>(
            ff1b, wb2, b2 + (size_t)i * D, x, nullptr, ROWS, D, FF);
    }

    layernorm_kernel<<<ROWS, 256, 0, stream>>>(x, lnf_s, lnf_b, hb);
    gemm_bf16_wmma<1, 0, 0, 0><<<dim3(V / 64, ROWS / 64), 256, 0, stream>>>(
        hb, wbout, bout, out, nullptr, ROWS, V, D);
}
